// Rerank_Module_72138270703657
// MI455X (gfx1250) — compile-verified, matrix operand order checked by reading, not run on hardware
//
#include <hip/hip_runtime.h>
#include <hip/hip_bf16.h>
#include <stdint.h>

// ---------------------------------------------------------------------------
// MI455X (gfx1250) implementation of the deformable-multiscale rerank module.
// All GEMMs run through v_wmma_f32_16x16x32_bf16 (bf16 in, fp32 accum), with
// the B tile staged into LDS by the Tensor Data Mover (TDM) and shared by all
// 8 waves of the workgroup.  Wave32-aware kernels handle top-k, bilinear
// sampling, softmax, layernorm.
// ---------------------------------------------------------------------------

typedef __bf16 bf16_t;
typedef __attribute__((ext_vector_type(16))) __bf16 v16bf;
typedef __attribute__((ext_vector_type(8)))  __bf16 v8bf;
typedef __attribute__((ext_vector_type(8)))  float  v8f;
typedef __attribute__((ext_vector_type(4)))  unsigned int uint32x4;
typedef __attribute__((ext_vector_type(8)))  int int32x8;
typedef __attribute__((ext_vector_type(4)))  int int32x4;

#define NT     384          // N_T tokens
#define KTOP   16
#define NQ     6144         // NT*KTOP
#define DMODEL 256
#define NHEAD  8
#define DH     32
#define DFF    1024
#define PROW   24576        // Hf*Wf
#define PTOT   32640        // sum of the 4 scales
#define WF     192

// output layout (floats): out[98304] | p_patch[12288] | u[6144] | s[6144]
#define OUT_PP 98304
#define OUT_U  110592
#define OUT_S  116736

// ---------------------------------------------------------------------------
// small helpers
// ---------------------------------------------------------------------------
__device__ __forceinline__ float wave_sum32(float v) {
#pragma unroll
  for (int o = 16; o > 0; o >>= 1) v += __shfl_xor(v, o, 32);
  return v;
}

__device__ __forceinline__ v16bf load_frag_g(const bf16_t* p) {
  union { v8bf h[2]; v16bf v; } u;
  u.h[0] = *reinterpret_cast<const v8bf*>(p);
  u.h[1] = *reinterpret_cast<const v8bf*>(p + 16);
  return u.v;
}

// ---------------------------------------------------------------------------
// elementwise / transpose kernels (one-time weight prep, casts)
// ---------------------------------------------------------------------------
__global__ __launch_bounds__(256) void transpose_cast(const float* __restrict__ src,
                                                      bf16_t* __restrict__ dst,
                                                      int R, int C) {
  int i = blockIdx.x * 256 + threadIdx.x;
  if (i >= R * C) return;
  int r = i / C, c = i - r * C;
  dst[(size_t)c * R + r] = (bf16_t)src[i];
}

__global__ __launch_bounds__(256) void cast_bf(const float* __restrict__ src,
                                               bf16_t* __restrict__ dst, int n) {
  int i = blockIdx.x * 256 + threadIdx.x;
  if (i < n) dst[i] = (bf16_t)src[i];
}

__global__ __launch_bounds__(256) void broadcast_q(const float* __restrict__ qt,
                                                   float* __restrict__ qf,
                                                   bf16_t* __restrict__ qb) {
  int i = blockIdx.x * 256 + threadIdx.x;
  if (i >= NQ * DMODEL) return;
  int row = i >> 8, c = i & 255;
  float v = qt[(size_t)(row >> 4) * DMODEL + c];   // row = token*16 + k
  qf[i] = v;
  qb[i] = (bf16_t)v;
}

__global__ __launch_bounds__(256) void build_cat(const float* __restrict__ left,
                                                 const float* __restrict__ qt,
                                                 bf16_t* __restrict__ out,
                                                 int rows, int repeat) {
  // out[row, 0:256] = left[row], out[row, 256:512] = qt[row/repeat]
  int i = blockIdx.x * 256 + threadIdx.x;
  if (i >= rows * DMODEL) return;
  int row = i >> 8, c = i & 255;
  out[(size_t)row * 512 + c]       = (bf16_t)left[i];
  out[(size_t)row * 512 + 256 + c] = (bf16_t)qt[(size_t)(row / repeat) * DMODEL + c];
}

// ---------------------------------------------------------------------------
// top-16 per score row (c_t row length 24576); emits p_patch + normalized ref
// ---------------------------------------------------------------------------
__global__ __launch_bounds__(256) void topk16(const float* __restrict__ c_t,
                                              float* __restrict__ refxy,
                                              float* __restrict__ p_patch,
                                              int ntok) {
  __shared__ float rmax[256];
  __shared__ int   rarg[256];
  __shared__ int   chosen[KTOP];
  int n = blockIdx.x;
  if (n >= ntok) return;
  int tid = threadIdx.x;
  const float* row = c_t + (size_t)n * PROW;
  for (int k = 0; k < KTOP; ++k) {
    float bm = -3.4e38f; int ba = 0;
    for (int i = tid; i < PROW; i += 256) {
      bool skip = false;
      for (int j = 0; j < k; ++j) skip |= (chosen[j] == i);
      float v = row[i];
      if (!skip && v > bm) { bm = v; ba = i; }
    }
    rmax[tid] = bm; rarg[tid] = ba;
    __syncthreads();
    for (int s = 128; s > 0; s >>= 1) {
      if (tid < s && rmax[tid + s] > rmax[tid]) { rmax[tid] = rmax[tid + s]; rarg[tid] = rarg[tid + s]; }
      __syncthreads();
    }
    if (tid == 0) {
      int win = rarg[0];
      chosen[k] = win;
      float px = (float)((win % WF) * 4 + 2);
      float py = (float)((win / WF) * 4 + 2);
      int nk = n * KTOP + k;
      p_patch[nk * 2]     = px;
      p_patch[nk * 2 + 1] = py;
      refxy[nk * 2]     = fminf(fmaxf(px / 768.f, 0.f), 1.f);
      refxy[nk * 2 + 1] = fminf(fmaxf(py / 512.f, 0.f), 1.f);
    }
    __syncthreads();
  }
}

// ---------------------------------------------------------------------------
// WMMA GEMM:  C[M,N] = A[M,K](bf16, row-major) * Bt[N,K](bf16, row-major)^T
// block = 8 waves, tile 128(M) x 64(N); B strip staged to LDS by the TDM.
// All 4 B fragments are loaded into distinct registers before the 4 WMMAs so
// the matrix ops issue back-to-back (single ds wait per k-step).
// epilogue: +bias, optional relu, fp32 and/or bf16 outputs.
// requires: M%128==0, N%64==0, K%256==0 (true for every call site here).
// ---------------------------------------------------------------------------
__global__ __launch_bounds__(256) void gemm_bf16(const bf16_t* __restrict__ A,
                                                 const bf16_t* __restrict__ Bt,
                                                 const float* __restrict__ bias,
                                                 float* __restrict__ C,
                                                 bf16_t* __restrict__ Cbf,
                                                 int M, int N, int Kd, int relu) {
  __shared__ bf16_t sB[64 * 256];      // 32 KB k-chunk of the 64-col B strip
  const int tid  = threadIdx.x;
  const int wave = tid >> 5;
  const int lane = tid & 31;
  const int l  = lane & 15;            // row-within-fragment (M for A, N for B)
  const int hi = lane >> 4;            // selects K sub-chunks {0,16} vs {8,24}
  const int m0 = blockIdx.x * 128 + wave * 16;
  const int n0 = blockIdx.y * 64;

  v8f acc[4];
#pragma unroll
  for (int t = 0; t < 4; ++t)
#pragma unroll
    for (int e = 0; e < 8; ++e) acc[t][e] = 0.f;

  const unsigned ldsb = (unsigned)(size_t)(void*)&sB[0];   // LDS byte offset

  for (int kb = 0; kb < Kd; kb += 256) {
    __syncthreads();                   // previous chunk fully consumed
    if (wave == 0) {
      // --- Tensor Data Mover: 2D tile (256 x 64) of Bt, elem=2B -------------
      unsigned long long ga =
          (unsigned long long)(uintptr_t)(Bt + (size_t)n0 * Kd + kb);
      uint32x4 g0;
      g0[0] = 1u;                                          // count=1, user D#
      g0[1] = ldsb;                                        // lds_addr
      g0[2] = (unsigned)(ga & 0xffffffffu);                // global_addr[31:0]
      g0[3] = (unsigned)((ga >> 32) & 0x1ffffffu) | (2u << 30); // [56:32]|type=2
      int32x8 g1;
      g1[0] = 0x10000;                                     // data_size=2B
      g1[1] = (int)((Kd & 0xffff) << 16);                  // tensor_dim0 lo
      g1[2] = (int)(((Kd >> 16) & 0xffff) | (64 << 16));   // dim0 hi | dim1 lo
      g1[3] = (int)(256 << 16);                            // tile_dim0 = 256
      g1[4] = 64;                                          // tile_dim1 = 64
      g1[5] = Kd;                                          // dim0_stride lo
      g1[6] = 0;
      g1[7] = 0;
      int32x4 gz; gz[0] = 0; gz[1] = 0; gz[2] = 0; gz[3] = 0;
      int32x8 gz8;
#pragma unroll
      for (int i = 0; i < 8; ++i) gz8[i] = 0;
      // 6-arg form (clang-23 / therock-10.0 headers): g0, g1, g2, g3, g4, cpol
      __builtin_amdgcn_tensor_load_to_lds(g0, g1, gz, gz, gz8, 0);
      __builtin_amdgcn_s_wait_tensorcnt(0);
    }
    __syncthreads();                   // chunk resident in LDS

#pragma unroll
    for (int ks = 0; ks < 256; ks += 32) {
      const bf16_t* pA = A + (size_t)(m0 + l) * Kd + (kb + ks + hi * 8);
      v16bf afr = load_frag_g(pA);
      v16bf bfr[4];
#pragma unroll
      for (int t = 0; t < 4; ++t) {
        const int bo = (t * 16 + l) * 256 + ks + hi * 8;
        union { v8bf h[2]; v16bf v; } ub;
        ub.h[0] = *reinterpret_cast<const v8bf*>(&sB[bo]);
        ub.h[1] = *reinterpret_cast<const v8bf*>(&sB[bo + 16]);
        bfr[t] = ub.v;
      }
#pragma unroll
      for (int t = 0; t < 4; ++t) {
        acc[t] = __builtin_amdgcn_wmma_f32_16x16x32_bf16(
            false, afr, false, bfr[t], (short)0, acc[t], false, false);
      }
    }
  }

#pragma unroll
  for (int t = 0; t < 4; ++t) {
    int n = n0 + t * 16 + l;
    float bv = bias ? bias[n] : 0.f;
#pragma unroll
    for (int e = 0; e < 8; ++e) {
      int m = m0 + e + hi * 8;         // C/D layout: vgpr e -> M=e (+8 hi half)
      float v = acc[t][e] + bv;
      if (relu) v = fmaxf(v, 0.f);
      size_t o = (size_t)m * N + n;
      if (C)   C[o]   = v;
      if (Cbf) Cbf[o] = (bf16_t)v;
    }
  }
}

// ---------------------------------------------------------------------------
// deformable multiscale sampling: one wave per (query, head); lane = channel
// ---------------------------------------------------------------------------
__global__ __launch_bounds__(256) void dms_sample(const float* __restrict__ value,
                                                  const float* __restrict__ off,
                                                  const float* __restrict__ awl,
                                                  const float* __restrict__ refxy,
                                                  bf16_t* __restrict__ outbf) {
  int t = blockIdx.x * 8 + (threadIdx.x >> 5);
  int lane = threadIdx.x & 31;
  int q = t >> 3, h = t & 7;
  // softmax over the 16 (level,point) logits of this head
  float w[16]; float mx = -3.4e38f;
#pragma unroll
  for (int j = 0; j < 16; ++j) { w[j] = awl[(size_t)q * 128 + h * 16 + j]; mx = fmaxf(mx, w[j]); }
  float ss = 0.f;
#pragma unroll
  for (int j = 0; j < 16; ++j) { w[j] = __expf(w[j] - mx); ss += w[j]; }
  float inv = 1.f / ss;
  float rx = refxy[q * 2], ry = refxy[q * 2 + 1];
  const int cH[4] = {128, 64, 32, 16};
  const int cW[4] = {192, 96, 48, 24};
  const int cS[4] = {0, 24576, 30720, 32256};
  float acc = 0.f;
#pragma unroll
  for (int lv = 0; lv < 4; ++lv) {
    int Hl = cH[lv], Wl = cW[lv], st = cS[lv];
#pragma unroll
    for (int pn = 0; pn < 4; ++pn) {
      int j = lv * 4 + pn;
      float ox = off[(size_t)q * 256 + h * 32 + lv * 8 + pn * 2];
      float oy = off[(size_t)q * 256 + h * 32 + lv * 8 + pn * 2 + 1];
      float x = (rx + ox / (float)Wl) * (float)Wl - 0.5f;
      float y = (ry + oy / (float)Hl) * (float)Hl - 0.5f;
      float x0f = floorf(x), y0f = floorf(y);
      float fx = x - x0f, fy = y - y0f;
      int x0 = (int)x0f, y0 = (int)y0f;
      float wj = w[j] * inv;
#pragma unroll
      for (int dy = 0; dy < 2; ++dy)
#pragma unroll
        for (int dx = 0; dx < 2; ++dx) {
          int xi = x0 + dx, yi = y0 + dy;
          if (xi >= 0 && xi < Wl && yi >= 0 && yi < Hl) {
            float wc = (dx ? fx : 1.f - fx) * (dy ? fy : 1.f - fy);
            acc += wj * wc * value[(size_t)(st + yi * Wl + xi) * 256 + h * 32 + lane];
          }
        }
    }
  }
  outbf[(size_t)q * 256 + h * 32 + lane] = (bf16_t)acc;
}

// ---------------------------------------------------------------------------
// fused residual-add + layernorm (wave per 256-wide row); writes fp32 + bf16
// ---------------------------------------------------------------------------
__global__ __launch_bounds__(256) void add_ln(const float* __restrict__ x,
                                              const float* __restrict__ r,
                                              const float* __restrict__ g,
                                              const float* __restrict__ b,
                                              float* __restrict__ y,
                                              bf16_t* __restrict__ ybf, int rows) {
  int row = blockIdx.x * 8 + (threadIdx.x >> 5);
  int lane = threadIdx.x & 31;
  if (row >= rows) return;
  float v[8]; float s = 0.f;
#pragma unroll
  for (int i = 0; i < 8; ++i) {
    int c = lane + i * 32;
    v[i] = x[(size_t)row * 256 + c] + r[(size_t)row * 256 + c];
    s += v[i];
  }
  s = wave_sum32(s);
  float mean = s * (1.f / 256.f);
  float var = 0.f;
#pragma unroll
  for (int i = 0; i < 8; ++i) { float d = v[i] - mean; var += d * d; }
  var = wave_sum32(var);
  float inv = rsqrtf(var * (1.f / 256.f) + 1e-5f);
#pragma unroll
  for (int i = 0; i < 8; ++i) {
    int c = lane + i * 32;
    float o = (v[i] - mean) * inv * g[c] + b[c];
    y[(size_t)row * 256 + c] = o;
    if (ybf) ybf[(size_t)row * 256 + c] = (bf16_t)o;
  }
}

// ---------------------------------------------------------------------------
// u = X . cert_W + cert_b ; s = X . score_W + score_b   (wave per row)
// ---------------------------------------------------------------------------
__global__ __launch_bounds__(256) void rowdot2(const float* __restrict__ X,
                                               const float* __restrict__ w1,
                                               const float* __restrict__ b1,
                                               const float* __restrict__ w2,
                                               const float* __restrict__ b2,
                                               float* __restrict__ u,
                                               float* __restrict__ s, int rows) {
  int row = blockIdx.x * 8 + (threadIdx.x >> 5);
  int lane = threadIdx.x & 31;
  if (row >= rows) return;
  float d1 = 0.f, d2 = 0.f;
#pragma unroll
  for (int i = 0; i < 8; ++i) {
    int c = lane + i * 32;
    float v = X[(size_t)row * 256 + c];
    d1 += v * w1[c]; d2 += v * w2[c];
  }
  d1 = wave_sum32(d1); d2 = wave_sum32(d2);
  if (lane == 0) { u[row] = d1 + b1[0]; s[row] = d2 + b2[0]; }
}

// ---------------------------------------------------------------------------
// MHA core: 1 query vs 16 keys, per (token, head) wave; lane = channel
// ---------------------------------------------------------------------------
__global__ __launch_bounds__(256) void mha_attn(const float* __restrict__ qh,
                                                const float* __restrict__ kh,
                                                const float* __restrict__ vh,
                                                bf16_t* __restrict__ obf) {
  int t = blockIdx.x * 8 + (threadIdx.x >> 5);
  int lane = threadIdx.x & 31;
  int n = t >> 3, h = t & 7;
  float qv = qh[(size_t)n * 256 + h * 32 + lane];
  float sc[16];
#pragma unroll
  for (int k = 0; k < 16; ++k) {
    float p = qv * kh[(size_t)(n * 16 + k) * 256 + h * 32 + lane];
    p = wave_sum32(p);
    sc[k] = p * 0.17677669529663689f;       // 1/sqrt(32)
  }
  float mx = -3.4e38f;
#pragma unroll
  for (int k = 0; k < 16; ++k) mx = fmaxf(mx, sc[k]);
  float ss = 0.f;
#pragma unroll
  for (int k = 0; k < 16; ++k) { sc[k] = __expf(sc[k] - mx); ss += sc[k]; }
  float inv = 1.f / ss;
  float acc = 0.f;
#pragma unroll
  for (int k = 0; k < 16; ++k)
    acc += sc[k] * inv * vh[(size_t)(n * 16 + k) * 256 + h * 32 + lane];
  obf[(size_t)n * 256 + h * 32 + lane] = (bf16_t)acc;
}

// ---------------------------------------------------------------------------
// host orchestration
// ---------------------------------------------------------------------------
extern "C" void kernel_launch(void* const* d_in, const int* in_sizes, int n_in,
                              void* d_out, int out_size, void* d_ws, size_t ws_size,
                              hipStream_t stream) {
  (void)in_sizes; (void)n_in; (void)out_size; (void)ws_size;
  // inputs in setup_inputs() order; params flattened in jax tree (alphabetical) order
  const float* q_t     = (const float*)d_in[0];   // [384,256]
  const float* f4      = (const float*)d_in[1];   // [24576,256]
  const float* f8      = (const float*)d_in[2];   // [6144,256]
  const float* f16p    = (const float*)d_in[3];   // [1536,256]
  const float* f32p    = (const float*)d_in[4];   // [384,256]
  const float* c_t     = (const float*)d_in[5];   // [384,24576]
  const float* cert_W  = (const float*)d_in[6];
  const float* cert_b  = (const float*)d_in[7];
  const float* dmWattn = (const float*)d_in[8];   // [3,256,128]
  const float* dmWf1   = (const float*)d_in[9];   // [3,256,1024]
  const float* dmWf2   = (const float*)d_in[10];  // [3,1024,256]
  const float* dmWoff  = (const float*)d_in[11];  // [3,256,256]
  const float* dmWout  = (const float*)d_in[12];  // [3,256,256]
  const float* dmWv    = (const float*)d_in[13];  // [3,256,256]
  const float* dmbattn = (const float*)d_in[14];
  const float* dmbf1   = (const float*)d_in[15];
  const float* dmbf2   = (const float*)d_in[16];
  const float* dmboff  = (const float*)d_in[17];
  const float* dmbout  = (const float*)d_in[18];
  const float* dmbv    = (const float*)d_in[19];
  const float* dmln1b  = (const float*)d_in[20];
  const float* dmln1g  = (const float*)d_in[21];
  const float* dmln2b  = (const float*)d_in[22];
  const float* dmln2g  = (const float*)d_in[23];
  const float* final_W = (const float*)d_in[24];  // [512,256]
  const float* final_b = (const float*)d_in[25];
  const float* fusion_W= (const float*)d_in[26];  // [512,256]
  const float* fusion_b= (const float*)d_in[27];
  const float* mWf1    = (const float*)d_in[28];  // [256,1024]
  const float* mWf2    = (const float*)d_in[29];  // [1024,256]
  const float* mWk     = (const float*)d_in[30];
  const float* mWo     = (const float*)d_in[31];
  const float* mWq     = (const float*)d_in[32];
  const float* mWv     = (const float*)d_in[33];
  const float* mbf1    = (const float*)d_in[34];
  const float* mbf2    = (const float*)d_in[35];
  const float* mbk     = (const float*)d_in[36];
  const float* mbo     = (const float*)d_in[37];
  const float* mbq     = (const float*)d_in[38];
  const float* mbv     = (const float*)d_in[39];
  const float* mln1b   = (const float*)d_in[40];
  const float* mln1g   = (const float*)d_in[41];
  const float* mln2b   = (const float*)d_in[42];
  const float* mln2g   = (const float*)d_in[43];
  const float* score_W = (const float*)d_in[44];
  const float* score_b = (const float*)d_in[45];

  float* out_f = (float*)d_out;

  // bump allocator on d_ws
  char* wsb = (char*)d_ws;
  size_t cur = 0;
  auto alloc = [&](size_t bytes) -> void* {
    void* p = wsb + cur;
    cur = (cur + bytes + 255) & ~(size_t)255;
    return p;
  };
  auto abf = [&](size_t elems) { return (bf16_t*)alloc(elems * sizeof(bf16_t)); };
  auto af32 = [&](size_t elems) { return (float*)alloc(elems * sizeof(float)); };

  bf16_t* fsb   = abf((size_t)PTOT * DMODEL);    // f_scales, bf16
  float*  val   = af32((size_t)PTOT * DMODEL);   // value (per layer)
  float*  qF    = af32((size_t)NQ * DMODEL);
  bf16_t* qB    = abf((size_t)NQ * DMODEL);
  float*  refxy = af32((size_t)NQ * 2);
  float*  offF  = af32((size_t)NQ * DMODEL);
  float*  awF   = af32((size_t)NQ * 128);
  bf16_t* attnB = abf((size_t)NQ * DMODEL);
  float*  tmpF  = af32((size_t)NQ * DMODEL);
  bf16_t* ff1B  = abf((size_t)NQ * DFF);
  bf16_t* qcatB = abf((size_t)NQ * 512);
  float*  qfF   = af32((size_t)NQ * DMODEL);
  bf16_t* qfB   = abf((size_t)NQ * DMODEL);
  bf16_t* qtB   = abf((size_t)NT * DMODEL);
  float*  qhF   = af32((size_t)NT * DMODEL);
  float*  khF   = af32((size_t)NQ * DMODEL);
  float*  vhF   = af32((size_t)NQ * DMODEL);
  bf16_t* oB    = abf((size_t)NT * DMODEL);
  float*  tmp2F = af32((size_t)NT * DMODEL);
  float*  xF    = af32((size_t)NT * DMODEL);
  bf16_t* xB    = abf((size_t)NT * DMODEL);
  bf16_t* ffmB  = abf((size_t)NT * DFF);
  float*  fusedF= af32((size_t)NT * DMODEL);
  bf16_t* cat2B = abf((size_t)NT * 512);

  // transposed bf16 weights (Bt layout [N,K])
  bf16_t *wvT[3], *woffT[3], *wattnT[3], *woutT[3], *wf1T[3], *wf2T[3];
  for (int l = 0; l < 3; ++l) {
    wvT[l]    = abf(65536);  woffT[l] = abf(65536);
    wattnT[l] = abf(32768);  woutT[l] = abf(65536);
    wf1T[l]   = abf(262144); wf2T[l]  = abf(262144);
  }
  bf16_t* wqT  = abf(65536);  bf16_t* wkT  = abf(65536);
  bf16_t* wvmT = abf(65536);  bf16_t* woT  = abf(65536);
  bf16_t* mf1T = abf(262144); bf16_t* mf2T = abf(262144);
  bf16_t* fusT = abf(131072); bf16_t* finT = abf(131072);

  auto tr = [&](const float* src, bf16_t* dst, int R, int C) {
    int n = R * C;
    transpose_cast<<<(n + 255) / 256, 256, 0, stream>>>(src, dst, R, C);
  };
  for (int l = 0; l < 3; ++l) {
    tr(dmWv   + (size_t)l * 65536,  wvT[l],    256, 256);
    tr(dmWoff + (size_t)l * 65536,  woffT[l],  256, 256);
    tr(dmWattn+ (size_t)l * 32768,  wattnT[l], 256, 128);
    tr(dmWout + (size_t)l * 65536,  woutT[l],  256, 256);
    tr(dmWf1  + (size_t)l * 262144, wf1T[l],   256, 1024);
    tr(dmWf2  + (size_t)l * 262144, wf2T[l],   1024, 256);
  }
  tr(mWq, wqT, 256, 256);   tr(mWk, wkT, 256, 256);
  tr(mWv, wvmT, 256, 256);  tr(mWo, woT, 256, 256);
  tr(mWf1, mf1T, 256, 1024); tr(mWf2, mf2T, 1024, 256);
  tr(fusion_W, fusT, 512, 256); tr(final_W, finT, 512, 256);

  // f_scales concat -> bf16
  cast_bf<<<(24576 * 256 + 255) / 256, 256, 0, stream>>>(f4,   fsb, 24576 * 256);
  cast_bf<<<(6144 * 256 + 255) / 256, 256, 0, stream>>>(f8,   fsb + (size_t)24576 * 256, 6144 * 256);
  cast_bf<<<(1536 * 256 + 255) / 256, 256, 0, stream>>>(f16p, fsb + (size_t)30720 * 256, 1536 * 256);
  cast_bf<<<(384 * 256 + 255) / 256, 256, 0, stream>>>(f32p, fsb + (size_t)32256 * 256, 384 * 256);
  cast_bf<<<(NT * 256 + 255) / 256, 256, 0, stream>>>(q_t, qtB, NT * 256);

  // top-16 selection + reference points (+ p_patch output)
  topk16<<<NT, 256, 0, stream>>>(c_t, refxy, out_f + OUT_PP, NT);
  broadcast_q<<<(NQ * DMODEL + 255) / 256, 256, 0, stream>>>(q_t, qF, qB);

  auto gemm = [&](const bf16_t* A, const bf16_t* Bt, const float* bias,
                  float* C, bf16_t* Cb, int M, int N, int Kd, int relu) {
    dim3 g(M / 128, N / 64);
    gemm_bf16<<<g, 256, 0, stream>>>(A, Bt, bias, C, Cb, M, N, Kd, relu);
  };

  // --- 3 deformable blocks ------------------------------------------------
  for (int l = 0; l < 3; ++l) {
    gemm(fsb, wvT[l], dmbv + l * 256, val, nullptr, PTOT, 256, 256, 0);
    gemm(qB, woffT[l],  dmboff  + l * 256, offF, nullptr, NQ, 256, 256, 0);
    gemm(qB, wattnT[l], dmbattn + l * 128, awF,  nullptr, NQ, 128, 256, 0);
    dms_sample<<<NQ, 256, 0, stream>>>(val, offF, awF, refxy, attnB);
    gemm(attnB, woutT[l], dmbout + l * 256, tmpF, nullptr, NQ, 256, 256, 0);
    add_ln<<<NQ / 8, 256, 0, stream>>>(qF, tmpF, dmln1g + l * 256, dmln1b + l * 256, qF, qB, NQ);
    gemm(qB, wf1T[l], dmbf1 + l * 1024, nullptr, ff1B, NQ, 1024, 256, 1);
    gemm(ff1B, wf2T[l], dmbf2 + l * 256, tmpF, nullptr, NQ, 256, 1024, 0);
    add_ln<<<NQ / 8, 256, 0, stream>>>(qF, tmpF, dmln2g + l * 256, dmln2b + l * 256, qF, qB, NQ);
  }

  // --- fusion + cert/score -----------------------------------------------
  build_cat<<<(NQ * DMODEL + 255) / 256, 256, 0, stream>>>(qF, q_t, qcatB, NQ, KTOP);
  gemm(qcatB, fusT, fusion_b, qfF, qfB, NQ, 256, 512, 0);
  rowdot2<<<NQ / 8, 256, 0, stream>>>(qfF, cert_W, cert_b, score_W, score_b,
                                      out_f + OUT_U, out_f + OUT_S, NQ);

  // --- MHA block ----------------------------------------------------------
  gemm(qtB, wqT,  mbq, qhF, nullptr, NT, 256, 256, 0);
  gemm(qfB, wkT,  mbk, khF, nullptr, NQ, 256, 256, 0);
  gemm(qfB, wvmT, mbv, vhF, nullptr, NQ, 256, 256, 0);
  mha_attn<<<NT, 256, 0, stream>>>(qhF, khF, vhF, oB);
  gemm(oB, woT, mbo, tmp2F, nullptr, NT, 256, 256, 0);
  add_ln<<<NT / 8, 256, 0, stream>>>(q_t, tmp2F, mln1g, mln1b, xF, xB, NT);
  gemm(xB, mf1T, mbf1, nullptr, ffmB, NT, 1024, 256, 1);
  gemm(ffmB, mf2T, mbf2, tmp2F, nullptr, NT, 256, 1024, 0);
  add_ln<<<NT / 8, 256, 0, stream>>>(xF, tmp2F, mln2g, mln2b, fusedF, nullptr, NT);
  build_cat<<<(NT * DMODEL + 255) / 256, 256, 0, stream>>>(fusedF, q_t, cat2B, NT, 1);
  gemm(cat2B, finT, final_b, out_f, nullptr, NT, 256, 512, 0);
}
